// ConvblockNofrills_64768106824121
// MI455X (gfx1250) — matrix operand order repaired and checked by reading, not compiled, not run on hardware
//
#include <hip/hip_runtime.h>

typedef __attribute__((ext_vector_type(16))) __bf16       v16bf;
typedef __attribute__((ext_vector_type(8)))  float        v8f;
typedef __attribute__((ext_vector_type(8)))  unsigned int v8u;
typedef __attribute__((ext_vector_type(4)))  unsigned int v4u;

#define T_DIM  4096
#define C_DIM  512
#define K_W    7
#define L_DIM  4090   // T - K + 1
#define NBATCH 8
#define TILE_M 64
#define TILE_N 64
#define PITCH  520    // bf16 elements per LDS row (512 + 8 pad -> bank-conflict-free)
#define XPITCH 68     // f32 elements per epilogue-x LDS row (64 + 4 pad)

__device__ __forceinline__ unsigned short f2bf(float f) {
  union { float f; unsigned u; } v; v.f = f;
  return (unsigned short)(v.u >> 16);          // truncate: exact hi for hi/lo split
}
__device__ __forceinline__ float bf2f(unsigned short h) {
  union { float f; unsigned u; } v; v.u = ((unsigned)h) << 16;
  return v.f;
}
// tanh(x) = 1 - 2/(exp(2x)+1); overflow-safe (exp2->inf => 1, exp2->0 => -1)
__device__ __forceinline__ float fast_tanh(float x) {
  float e = __builtin_amdgcn_exp2f(x * 2.8853900817779268f);  // 2*log2(e)
  return 1.0f - 2.0f / (e + 1.0f);
}

// ---------------------------------------------------------------------------
// TDM: async 2D tile load  global(bf16 rows, row stride 512 elem) -> LDS,
// padding 4 dwords every 256 dwords so LDS row pitch becomes 520 bf16.
// D# per CDNA5 ISA ch.8: group0 (4 SGPRs) + group1 (8 SGPRs), 2D tile.
// ---------------------------------------------------------------------------
__device__ __forceinline__ void tdm_load_tile_2d(unsigned lds_off,
                                                 const unsigned short* gbase,
                                                 unsigned rows_avail) {
  unsigned long long ga = (unsigned long long)(size_t)gbase;
  v4u g0; v8u g1;
  g0[0] = 1u;                                            // count=1, user mode
  g0[1] = lds_off;                                       // lds_addr (bytes)
  g0[2] = (unsigned)ga;                                  // global_addr[31:0]
  g0[3] = (unsigned)((ga >> 32) & 0x01FFFFFFull)         // global_addr[56:32]
        | (2u << 30);                                    // type = 2 ("image")
  g1[0] = (1u << 16)                                     // data_size = 2 bytes
        | (1u << 20)                                     // pad_enable
        | (7u << 22)                                     // pad_interval = 256 dwords
        | (3u << 25);                                    // pad_amount   = 4 dwords
  g1[1] = ((unsigned)C_DIM & 0xFFFFu) << 16;             // tensor_dim0 lo16 = 512
  g1[2] = ((unsigned)C_DIM >> 16)                        // tensor_dim0 hi16
        | ((rows_avail & 0xFFFFu) << 16);                // tensor_dim1 lo16
  g1[3] = (rows_avail >> 16)                             // tensor_dim1 hi16
        | ((unsigned)C_DIM << 16);                       // tile_dim0 = 512
  g1[4] = 64u;                                           // tile_dim1 = 64, tile_dim2 = 0
  g1[5] = (unsigned)C_DIM;                               // tensor_dim0_stride lo32 = 512
  g1[6] = 0u;                                            // stride hi / dim1_stride (unused, 2D)
  g1[7] = 0u;
  asm volatile("tensor_load_to_lds %0, %1" :: "s"(g0), "s"(g1) : "memory");
}

// ---------------------------------------------------------------------------
// Prep kernel 1: split weights (D,C,K) fp32 into bf16 hi/lo, transposed to
// [k][d][c] so each (k, d-block) is a contiguous 64x512 bf16 tile for TDM.
// ---------------------------------------------------------------------------
__global__ __launch_bounds__(256)
void prep_weights(const float* __restrict__ wts,
                  unsigned short* __restrict__ wh,
                  unsigned short* __restrict__ wl) {
  int idx = blockIdx.x * 256 + threadIdx.x;       // (d,c) pair, 512*512 total
  int d = idx >> 9, c = idx & 511;
  const float* src = wts + (size_t)idx * K_W;     // (d*512 + c)*7
  #pragma unroll
  for (int k = 0; k < K_W; ++k) {
    float f = src[k];
    unsigned short h  = f2bf(f);
    unsigned short lo = f2bf(f - bf2f(h));
    size_t o = ((size_t)(k * C_DIM + d)) * C_DIM + c;
    wh[o] = h;
    wl[o] = lo;
  }
}

// ---------------------------------------------------------------------------
// Prep kernel 2: split x (B,T,C) fp32 into bf16 hi/lo arrays (same layout).
// ---------------------------------------------------------------------------
__global__ __launch_bounds__(256)
void prep_x(const float4* __restrict__ x4,
            unsigned* __restrict__ xh32,
            unsigned* __restrict__ xl32) {
  size_t i = (size_t)blockIdx.x * 256 + threadIdx.x;   // float4 index
  float4 f = x4[i];
  unsigned short hx = f2bf(f.x), hy = f2bf(f.y), hz = f2bf(f.z), hw = f2bf(f.w);
  unsigned short lx = f2bf(f.x - bf2f(hx)), ly = f2bf(f.y - bf2f(hy));
  unsigned short lz = f2bf(f.z - bf2f(hz)), lw = f2bf(f.w - bf2f(hw));
  xh32[2 * i]     = (unsigned)hx | ((unsigned)hy << 16);
  xh32[2 * i + 1] = (unsigned)hz | ((unsigned)hw << 16);
  xl32[2 * i]     = (unsigned)lx | ((unsigned)ly << 16);
  xl32[2 * i + 1] = (unsigned)lz | ((unsigned)lw << 16);
}

// ---------------------------------------------------------------------------
// Main kernel: per (l-tile, d-tile, batch) workgroup.
//   A panel  (64 l-rows x 512 c, bf16 hi/lo) : TDM once
//   B_k tile (64 d-rows x 512 c, bf16 hi/lo) : TDM per k
//   GEMM: v_wmma_f32_16x16x32_bf16, split-bf16 (hi*hi + hi*lo + lo*hi)
//   Fused epilogue: out += tanh(w) * x[b, l+k, d]
// ---------------------------------------------------------------------------
__global__ __launch_bounds__(256)
void convblock_dynconv_wmma(const float* __restrict__ x,
                            const unsigned short* __restrict__ xh,
                            const unsigned short* __restrict__ xl,
                            const unsigned short* __restrict__ wh,
                            const unsigned short* __restrict__ wl,
                            float* __restrict__ out)
{
  extern __shared__ char smem[];
  unsigned short* sAh = (unsigned short*)smem;            // A hi   64 x PITCH bf16
  unsigned short* sAl = sAh + TILE_M * PITCH;             // A lo
  unsigned short* sBh = sAl + TILE_M * PITCH;             // B hi   64 x PITCH bf16
  unsigned short* sBl = sBh + TILE_N * PITCH;             // B lo
  float*          sX  = (float*)(sBl + TILE_N * PITCH);   // 70 x XPITCH f32

  const int t     = threadIdx.x;
  const int lbase = blockIdx.x * TILE_M;
  const int dbase = blockIdx.y * TILE_N;
  const int bb    = blockIdx.z;

  // ---- async-stage A panel (rows lbase+6 .. lbase+69) via TDM (wave 0 only)
  if (t < 32) {
    unsigned rowsA = (unsigned)(T_DIM - (lbase + 6));     // OOB rows read as zero
    size_t aoff = ((size_t)(bb * T_DIM + lbase + 6)) * C_DIM;
    tdm_load_tile_2d((unsigned)(size_t)sAh, xh + aoff, rowsA);
    tdm_load_tile_2d((unsigned)(size_t)sAl, xl + aoff, rowsA);
  }

  // ---- stage epilogue x block: rows lbase..lbase+69, cols dbase..dbase+63 (fp32)
  for (int lin = t; lin < 70 * 64; lin += 256) {
    int r = lin >> 6, n = lin & 63;
    int gl = lbase + r; gl = gl > (T_DIM - 1) ? (T_DIM - 1) : gl;
    sX[r * XPITCH + n] = x[((size_t)bb * T_DIM + gl) * C_DIM + dbase + n];
  }

  const int lane = t & 31;
  const int w    = t >> 5;
  const int m0   = (w & 3) * 16;   // 4 waves along M
  const int n0   = (w >> 2) * 32;  // 2 waves along N, each does 2 N-tiles
  const int Ml   = lane & 15;      // M (A) / N (B,D) lane index
  const int hiL  = lane >> 4;      // lane half

  float outacc[2][8];
  #pragma unroll
  for (int nt = 0; nt < 2; ++nt)
    #pragma unroll
    for (int j = 0; j < 8; ++j) outacc[nt][j] = 0.f;

  const unsigned* Ah32 = (const unsigned*)sAh;
  const unsigned* Al32 = (const unsigned*)sAl;
  const unsigned* Bh32 = (const unsigned*)sBh;
  const unsigned* Bl32 = (const unsigned*)sBl;

  for (int k = 0; k < K_W; ++k) {
    // ---- async-stage B_k (64 d-rows x 512 c) via TDM (wave 0 only)
    if (t < 32) {
      size_t boff = ((size_t)(k * C_DIM + dbase)) * C_DIM;
      tdm_load_tile_2d((unsigned)(size_t)sBh, wh + boff, 64u);
      tdm_load_tile_2d((unsigned)(size_t)sBl, wl + boff, 64u);
      __builtin_amdgcn_s_wait_tensorcnt(0);
    }
    __syncthreads();

    v8f acc0 = {}; v8f acc1 = {};
    #pragma unroll 2
    for (int cc = 0; cc < C_DIM; cc += 32) {
      v8u ah, al, bh0, bl0, bh1, bl1;
      #pragma unroll
      for (int v = 0; v < 8; ++v) {
        // A 16x32 bf16 fragment (ISA 7.12.2): lane=M; K = (v&3)*2 + hiL*8 + (v>>2)*16
        int ka = ((v & 3) << 1) + (hiL << 3) + ((v >> 2) << 4);
        int ea = (m0 + Ml) * PITCH + cc + ka;
        ah[v] = Ah32[ea >> 1];
        al[v] = Al32[ea >> 1];
        // B 32x16 bf16 fragment: lane=N; K = (v&3)*2 + hiL*8 + (v>>2)*16 (same as A)
        int kb  = ((v & 3) << 1) + (hiL << 3) + ((v >> 2) << 4);
        int eb0 = (n0 + Ml) * PITCH + cc + kb;
        int eb1 = (n0 + 16 + Ml) * PITCH + cc + kb;
        bh0[v] = Bh32[eb0 >> 1];  bl0[v] = Bl32[eb0 >> 1];
        bh1[v] = Bh32[eb1 >> 1];  bl1[v] = Bl32[eb1 >> 1];
      }
      v16bf A_h = __builtin_bit_cast(v16bf, ah);
      v16bf A_l = __builtin_bit_cast(v16bf, al);
      v16bf B0h = __builtin_bit_cast(v16bf, bh0);
      v16bf B0l = __builtin_bit_cast(v16bf, bl0);
      v16bf B1h = __builtin_bit_cast(v16bf, bh1);
      v16bf B1l = __builtin_bit_cast(v16bf, bl1);
      // split-bf16 fp32 emulation: hi*hi + hi*lo + lo*hi
      acc0 = __builtin_amdgcn_wmma_f32_16x16x32_bf16(false, A_h, false, B0h, (short)0, acc0, false, false);
      acc1 = __builtin_amdgcn_wmma_f32_16x16x32_bf16(false, A_h, false, B1h, (short)0, acc1, false, false);
      acc0 = __builtin_amdgcn_wmma_f32_16x16x32_bf16(false, A_h, false, B0l, (short)0, acc0, false, false);
      acc1 = __builtin_amdgcn_wmma_f32_16x16x32_bf16(false, A_h, false, B1l, (short)0, acc1, false, false);
      acc0 = __builtin_amdgcn_wmma_f32_16x16x32_bf16(false, A_l, false, B0h, (short)0, acc0, false, false);
      acc1 = __builtin_amdgcn_wmma_f32_16x16x32_bf16(false, A_l, false, B1h, (short)0, acc1, false, false);
    }

    // ---- fused epilogue for this k: out += tanh(w) * x[b, l+k, d]
    #pragma unroll
    for (int j = 0; j < 8; ++j) {
      int mrow = m0 + j + hiL * 8 + k;        // row in sX (<= 69)
      outacc[0][j] += fast_tanh(acc0[j]) * sX[mrow * XPITCH + n0 + Ml];
      outacc[1][j] += fast_tanh(acc1[j]) * sX[mrow * XPITCH + n0 + 16 + Ml];
    }
    __syncthreads();   // before next k overwrites sB
  }

  // ---- store (B, L, C) with masking of the partial last l-tile
  #pragma unroll
  for (int j = 0; j < 8; ++j) {
    int l = lbase + m0 + j + hiL * 8;
    if (l < L_DIM) {
      size_t off = ((size_t)bb * L_DIM + l) * C_DIM + dbase;
      out[off + n0 + Ml]      = outacc[0][j];
      out[off + n0 + 16 + Ml] = outacc[1][j];
    }
  }
}

extern "C" void kernel_launch(void* const* d_in, const int* in_sizes, int n_in,
                              void* d_out, int out_size, void* d_ws, size_t ws_size,
                              hipStream_t stream) {
  (void)in_sizes; (void)n_in; (void)out_size; (void)ws_size;
  const float* x   = (const float*)d_in[0];
  const float* wts = (const float*)d_in[1];
  float*       out = (float*)d_out;

  // workspace layout: xh | xl | wh | wl  (bf16 hi/lo splits), ~71 MB total
  const size_t xelems = (size_t)NBATCH * T_DIM * C_DIM;        // 16,777,216
  const size_t welems = (size_t)C_DIM * C_DIM * K_W;           //  1,835,008
  unsigned short* xh = (unsigned short*)d_ws;
  unsigned short* xl = xh + xelems;
  unsigned short* wh = xl + xelems;
  unsigned short* wl = wh + welems;

  // prep: one-time fp32 -> bf16 hi/lo splits (memory-bound, ~6 us)
  hipLaunchKernelGGL(prep_x, dim3((unsigned)(xelems / 4 / 256)), dim3(256), 0, stream,
                     (const float4*)x, (unsigned*)xh, (unsigned*)xl);
  hipLaunchKernelGGL(prep_weights, dim3((unsigned)(C_DIM * C_DIM / 256)), dim3(256), 0, stream,
                     wts, wh, wl);

  // main fused GEMM + tanh + shifted-window reduction
  dim3 grid((L_DIM + TILE_M - 1) / TILE_M,   // 64 l-tiles
            C_DIM / TILE_N,                  // 8 d-tiles
            NBATCH);                         // batches
  dim3 block(256);
  size_t shmem = (size_t)(4 * TILE_M * PITCH * sizeof(unsigned short))  // A/B hi+lo
               + (size_t)(70 * XPITCH * sizeof(float));                 // epilogue x
  hipLaunchKernelGGL(convblock_dynconv_wmma, grid, block, shmem, stream,
                     x, xh, xl, wh, wl, out);
}
